// SpiralAutoencoder_LSTM_63711544868974
// MI455X (gfx1250) — matrix operand order repaired and checked by reading, not compiled, not run on hardware
//
#include <hip/hip_runtime.h>
#include <hip/hip_bf16.h>
#include <math.h>

// ---------------------------------------------------------------------------
// Types / WMMA helpers (CDNA5 gfx1250, wave32)
// ---------------------------------------------------------------------------
typedef __attribute__((ext_vector_type(16))) _Float16 v16h;
typedef __attribute__((ext_vector_type(8)))  float    v8f;

__device__ __forceinline__ v8f wmma_f16(v16h a, v16h b, v8f c) {
    // D = A(16x32 f16) x B(32x16 f16) + C(16x16 f32)
    return __builtin_amdgcn_wmma_f32_16x16x32_f16(false, a, false, b,
                                                  (short)0, c, false, false);
}

__device__ __forceinline__ float sigf(float x) { return 1.f / (1.f + __expf(-x)); }

__device__ __forceinline__ v8f v8f_splat(float x) {
    v8f v;
#pragma unroll
    for (int r = 0; r < 8; ++r) v[r] = x;
    return v;
}

// A-matrix fragment (16x32 f16): lane<16 -> row M=lane, K={0..7,16..23};
// lane>=16 -> same row, K={8..15,24..31}. p points at row start + k-chunk base.
__device__ __forceinline__ v16h load_frag_a(const _Float16* __restrict__ p, int hi) {
    v16h r;
    const int o = hi * 8;
#pragma unroll
    for (int j = 0; j < 8; ++j) {
        r[j]     = p[o + j];
        r[8 + j] = p[16 + o + j];
    }
    return r;
}

// B-matrix fragment (32x16 f16): lane holds column N=lane%16; lane<16 -> K={0..7,16..23},
// lane>=16 -> K={8..15,24..31}. p points at column-row start + k-chunk base.
__device__ __forceinline__ v16h load_frag_b(const _Float16* __restrict__ p, int hi) {
    v16h r;
    const int o = hi * 8;
#pragma unroll
    for (int j = 0; j < 8; ++j) {
        r[j]     = p[o + j];
        r[8 + j] = p[16 + o + j];
    }
    return r;
}

// ---------------------------------------------------------------------------
// Generic 16x16-tile WMMA GEMM:  Out[b] = A[b] (MxK) * BT[b]^T (KxN) + bias
//   A row-major (ldA), BT row-major (N rows of K, ldB), Out row-major f16 (ldO).
//   One wave per 16x16 output tile, 4 waves per block.
// ---------------------------------------------------------------------------
__global__ __launch_bounds__(128) void gemm16_f16(
    const _Float16* __restrict__ A, const _Float16* __restrict__ BT,
    _Float16* __restrict__ Out, const float* __restrict__ bias,
    int K, int ldA, int ldB, int ldO,
    long long sA, long long sB, long long sO,
    int Mt, int Nt, int nTiles)
{
    const int wave = threadIdx.x >> 5;
    const int lane = threadIdx.x & 31;
    const int tile = blockIdx.x * 4 + wave;
    if (tile >= nTiles) return;

    const int perB = Mt * Nt;
    const int bidx = tile / perB;
    const int rem  = tile % perB;
    const int tm = rem / Nt, tn = rem % Nt;
    const int m  = lane & 15;
    const int hi = lane >> 4;

    const _Float16* Ab = A  + (size_t)bidx * sA + (size_t)(tm * 16 + m) * ldA;
    const _Float16* Bb = BT + (size_t)bidx * sB + (size_t)(tn * 16 + m) * ldB;

    v8f acc = v8f_splat(bias ? bias[tn * 16 + m] : 0.f);

    for (int kb = 0; kb < K; kb += 32) {
        __builtin_prefetch(Ab + kb + 128, 0, 1);   // global_prefetch_b8
        __builtin_prefetch(Bb + kb + 128, 0, 1);
        v16h af = load_frag_a(Ab + kb, hi);
        v16h bf = load_frag_b(Bb + kb, hi);
        acc = wmma_f16(af, bf, acc);
    }

    _Float16* Ob = Out + (size_t)bidx * sO;
#pragma unroll
    for (int r = 0; r < 8; ++r) {
        const int row = tm * 16 + r + hi * 8;
        const int col = tn * 16 + m;
        Ob[(size_t)row * ldO + col] = (_Float16)acc[r];
    }
}

// ---------------------------------------------------------------------------
// Spiral-LSTM kernel. One wave owns a 16-sequence M-tile (fixed batch b,
// vertices n0..n0+15). Per step: gather x via spiral indices (row-major f16
// input, CPAD cols), WMMA gate GEMMs (x@Wih^T + h@Whh^T + b), elementwise cell
// update in accumulator layout, h kept in LDS (f16, A-source), c in registers.
// Output: elu(h) * dummy-vertex mask, written TRANSPOSED as [b][ch][vertex].
// ---------------------------------------------------------------------------
template <int CPAD, int HPAD, int H>
__global__ __launch_bounds__(128) void lstm_kernel(
    const _Float16* __restrict__ Xin,   // [B][Nv][CPAD] (pad cols zeroed)
    _Float16* __restrict__ xTout,       // [B][H][Nv]
    const int* __restrict__ spiral,     // [Nv][16]
    const _Float16* __restrict__ Wih,   // [4H][CPAD] zero-padded f16
    const _Float16* __restrict__ Whh,   // [4H][HPAD] zero-padded f16
    const float* __restrict__ bsum,     // [4H] = bih + bhh
    int Nv)
{
    __shared__ _Float16 hT[4][16 * HPAD];

    const int wave = threadIdx.x >> 5;
    const int lane = threadIdx.x & 31;
    const int tile = blockIdx.x * 4 + wave;        // grid is an exact multiple
    const int tilesPerB = Nv / 16;
    const int b  = tile / tilesPerB;
    const int n0 = (tile % tilesPerB) * 16;
    const int m  = lane & 15;
    const int hi = lane >> 4;

    _Float16* ht = hT[wave];
    for (int i = lane; i < 16 * HPAD; i += 32) ht[i] = (_Float16)0.f;

    int sidx[16];
#pragma unroll
    for (int t = 0; t < 16; ++t) sidx[t] = spiral[(n0 + m) * 16 + t];

    v8f cst[H / 16];
#pragma unroll
    for (int j = 0; j < H / 16; ++j) cst[j] = v8f_splat(0.f);

    __syncthreads();

    const size_t xbase = (size_t)b * Nv * CPAD;

    for (int t = 0; t < 16; ++t) {
        // x A-fragments for this timestep (gathered row)
        v16h xf[CPAD / 32];
        {
            const _Float16* row = Xin + xbase + (size_t)sidx[t] * CPAD;
#pragma unroll
            for (int kc = 0; kc < CPAD / 32; ++kc)
                xf[kc] = load_frag_a(row + kc * 32, hi);
        }
        // h A-fragments from LDS (previous step's h)
        v16h hf[HPAD / 32];
#pragma unroll
        for (int kc = 0; kc < HPAD / 32; ++kc)
            hf[kc] = load_frag_a(ht + m * HPAD + kc * 32, hi);

#pragma unroll
        for (int jc = 0; jc < H / 16; ++jc) {
            const int col = jc * 16 + m;           // hidden index (gate column)
            v8f acc[4];
#pragma unroll
            for (int q = 0; q < 4; ++q) acc[q] = v8f_splat(bsum[q * H + col]);

#pragma unroll
            for (int kc = 0; kc < CPAD / 32; ++kc) {
#pragma unroll
                for (int q = 0; q < 4; ++q) {
                    v16h bf = load_frag_b(Wih + (size_t)(q * H + col) * CPAD + kc * 32, hi);
                    acc[q] = wmma_f16(xf[kc], bf, acc[q]);
                }
            }
#pragma unroll
            for (int kc = 0; kc < HPAD / 32; ++kc) {
#pragma unroll
                for (int q = 0; q < 4; ++q) {
                    v16h bf = load_frag_b(Whh + (size_t)(q * H + col) * HPAD + kc * 32, hi);
                    acc[q] = wmma_f16(hf[kc], bf, acc[q]);
                }
            }
            // elementwise cell update: gate order i,f,g,o
#pragma unroll
            for (int r = 0; r < 8; ++r) {
                const float ig = sigf(acc[0][r]);
                const float fg = sigf(acc[1][r]);
                const float gg = tanhf(acc[2][r]);
                const float og = sigf(acc[3][r]);
                float cv = fg * cst[jc][r] + ig * gg;
                cst[jc][r] = cv;
                const float hv = og * tanhf(cv);
                const int row = r + hi * 8;
                ht[row * HPAD + col] = (_Float16)hv;
            }
        }
        __syncthreads();   // h visible for next step (and final output stage)
    }

    // output: elu + dummy-vertex mask, transposed store [b][j][vertex]
    for (int e = lane; e < 16 * H; e += 32) {
        const int mm = e & 15;
        const int j  = e >> 4;
        float v = (float)ht[mm * HPAD + j];
        v = v > 0.f ? v : expm1f(v);
        if (n0 + mm == Nv - 1) v = 0.f;
        xTout[((size_t)b * H + j) * Nv + n0 + mm] = (_Float16)v;
    }
}

// ---------------------------------------------------------------------------
// Small utility kernels
// ---------------------------------------------------------------------------
__global__ void cvt_pad_kernel(const float* __restrict__ src, _Float16* __restrict__ dst,
                               long long tot, int Cs, int Cd) {
    long long i = (long long)blockIdx.x * blockDim.x + threadIdx.x;
    if (i >= tot) return;
    const int c = (int)(i % Cd);
    const long long r = i / Cd;
    const float v = (c < Cs) ? src[r * Cs + c] : 0.f;
    dst[i] = (_Float16)v;
}

__global__ void vecadd_kernel(const float* __restrict__ a, const float* __restrict__ b,
                              float* __restrict__ o, int n) {
    int i = blockIdx.x * blockDim.x + threadIdx.x;
    if (i < n) o[i] = a[i] + b[i];
}

// [b][m][C] f16 -> [b][c][M] f16
__global__ void transpose_bmc_kernel(const _Float16* __restrict__ in, _Float16* __restrict__ out,
                                     int Mv, int C, long long tot) {
    long long i = (long long)blockIdx.x * blockDim.x + threadIdx.x;
    if (i >= tot) return;
    const int c = (int)(i % C);
    const long long t = i / C;
    const int m = (int)(t % Mv);
    const long long b = t / Mv;
    out[(b * C + c) * Mv + m] = in[i];
}

// final linear head: out[b][v][o] = sum_c xT[b][c][v] * W[o][c] + bias[o]
__global__ void lin_kernel(const _Float16* __restrict__ xT, const float* __restrict__ W,
                           const float* __restrict__ bias, float* __restrict__ out,
                           int Nv, long long tot) {
    long long i = (long long)blockIdx.x * blockDim.x + threadIdx.x;
    if (i >= tot) return;
    const int v = (int)(i % Nv);
    const long long b = i / Nv;
    float a0 = bias[0], a1 = bias[1], a2 = bias[2];
#pragma unroll
    for (int c = 0; c < 16; ++c) {
        const float x = (float)xT[(b * 16 + c) * (long long)Nv + v];
        a0 += x * W[c];
        a1 += x * W[16 + c];
        a2 += x * W[32 + c];
    }
    float* o = out + i * 3;
    o[0] = a0; o[1] = a1; o[2] = a2;
}

// ---------------------------------------------------------------------------
// Host orchestration
// ---------------------------------------------------------------------------
static const int kNS[5]    = {8192, 2048, 512, 128, 32};
static const int kEncC[4]  = {3, 16, 32, 64};
static const int kEncH[4]  = {16, 32, 64, 128};
static const int kDecC[4]  = {128, 64, 32, 32};
static const int kDecH[4]  = {64, 32, 32, 16};
static const int kEncCP[4] = {32, 32, 32, 64};
static const int kEncHP[4] = {32, 32, 64, 128};
static const int kDecCP[4] = {128, 64, 32, 32};
static const int kDecHP[4] = {64, 32, 32, 32};
static const int kB = 32;

extern "C" void kernel_launch(void* const* d_in, const int* in_sizes, int n_in,
                              void* d_out, int out_size, void* d_ws, size_t ws_size,
                              hipStream_t stream) {
    (void)in_sizes; (void)out_size; (void)ws_size;

    // ---- resolve input pointers (51 separate, or tuple-concatenated) ----
    const void* p[51];
    if (n_in >= 51) {
        for (int i = 0; i < 51; ++i) p[i] = d_in[i];
    } else {
        int j = 0;
        p[0] = d_in[j++];
        for (int i = 0; i < 4; ++i) p[1 + i] = d_in[j++];
        { const float* b = (const float*)d_in[j++]; size_t o = 0;
          for (int i = 0; i < 4; ++i) { p[5 + i] = b + o; o += (size_t)kNS[i + 1] * kNS[i]; } }
        { const float* b = (const float*)d_in[j++]; size_t o = 0;
          for (int i = 0; i < 4; ++i) { p[9 + i] = b + o; o += (size_t)kNS[i] * kNS[i + 1]; } }
        { const float* b = (const float*)d_in[j++]; size_t o = 0;
          for (int i = 0; i < 4; ++i) { p[13 + i] = b + o; o += (size_t)4 * kEncH[i] * kEncC[i]; } }
        { const float* b = (const float*)d_in[j++]; size_t o = 0;
          for (int i = 0; i < 4; ++i) { p[17 + i] = b + o; o += (size_t)4 * kEncH[i] * kEncH[i]; } }
        { const float* b = (const float*)d_in[j++]; size_t o = 0;
          for (int i = 0; i < 4; ++i) { p[21 + i] = b + o; o += (size_t)4 * kEncH[i]; } }
        { const float* b = (const float*)d_in[j++]; size_t o = 0;
          for (int i = 0; i < 4; ++i) { p[25 + i] = b + o; o += (size_t)4 * kEncH[i]; } }
        { const float* b = (const float*)d_in[j++]; size_t o = 0;
          for (int i = 0; i < 4; ++i) { p[29 + i] = b + o; o += (size_t)4 * kDecH[i] * kDecC[i]; } }
        { const float* b = (const float*)d_in[j++]; size_t o = 0;
          for (int i = 0; i < 4; ++i) { p[33 + i] = b + o; o += (size_t)4 * kDecH[i] * kDecH[i]; } }
        { const float* b = (const float*)d_in[j++]; size_t o = 0;
          for (int i = 0; i < 4; ++i) { p[37 + i] = b + o; o += (size_t)4 * kDecH[i]; } }
        { const float* b = (const float*)d_in[j++]; size_t o = 0;
          for (int i = 0; i < 4; ++i) { p[41 + i] = b + o; o += (size_t)4 * kDecH[i]; } }
        for (int i = 0; i < 6; ++i) p[45 + i] = d_in[j++];
    }

    const float* xin = (const float*)p[0];
    const int* sp[4] = {(const int*)p[1], (const int*)p[2], (const int*)p[3], (const int*)p[4]};
    const float *Din[4], *Uin[4], *eWih[4], *eWhh[4], *eBih[4], *eBhh[4];
    const float *dWih[4], *dWhh[4], *dBih[4], *dBhh[4];
    for (int i = 0; i < 4; ++i) {
        Din[i]  = (const float*)p[5 + i];   Uin[i]  = (const float*)p[9 + i];
        eWih[i] = (const float*)p[13 + i];  eWhh[i] = (const float*)p[17 + i];
        eBih[i] = (const float*)p[21 + i];  eBhh[i] = (const float*)p[25 + i];
        dWih[i] = (const float*)p[29 + i];  dWhh[i] = (const float*)p[33 + i];
        dBih[i] = (const float*)p[37 + i];  dBhh[i] = (const float*)p[41 + i];
    }
    const float* fceW = (const float*)p[45]; const float* fceB = (const float*)p[46];
    const float* fcdW = (const float*)p[47]; const float* fcdB = (const float*)p[48];
    const float* linW = (const float*)p[49]; const float* linB = (const float*)p[50];

    // ---- workspace bump allocator ----
    char* wsb = (char*)d_ws;
    size_t off = 0;
    auto alloc_h = [&](size_t n) -> _Float16* {
        size_t a = (off + 255) & ~(size_t)255; _Float16* r = (_Float16*)(wsb + a);
        off = a + n * sizeof(_Float16); return r; };
    auto alloc_f = [&](size_t n) -> float* {
        size_t a = (off + 255) & ~(size_t)255; float* r = (float*)(wsb + a);
        off = a + n * sizeof(float); return r; };

    // weights (f16, padded)
    _Float16 *DwH[4], *UwH[4], *eWihP[4], *eWhhP[4], *dWihP[4], *dWhhP[4];
    float *eBs[4], *dBs[4];
    for (int i = 0; i < 4; ++i) DwH[i] = alloc_h((size_t)kNS[i + 1] * kNS[i]);
    for (int i = 0; i < 4; ++i) UwH[i] = alloc_h((size_t)kNS[i] * kNS[i + 1]);
    _Float16* fceWH = alloc_h((size_t)256 * 4096);
    _Float16* fcdWH = alloc_h((size_t)4096 * 256);
    for (int i = 0; i < 4; ++i) {
        eWihP[i] = alloc_h((size_t)4 * kEncH[i] * kEncCP[i]);
        eWhhP[i] = alloc_h((size_t)4 * kEncH[i] * kEncHP[i]);
        eBs[i]   = alloc_f((size_t)4 * kEncH[i]);
        dWihP[i] = alloc_h((size_t)4 * kDecH[i] * kDecCP[i]);
        dWhhP[i] = alloc_h((size_t)4 * kDecH[i] * kDecHP[i]);
        dBs[i]   = alloc_f((size_t)4 * kDecH[i]);
    }
    // activations
    _Float16* xe0 = alloc_h((size_t)kB * 8192 * 32);
    _Float16* he0 = alloc_h((size_t)kB * 16 * 8192);
    _Float16* xe1 = alloc_h((size_t)kB * 2048 * 32);
    _Float16* he1 = alloc_h((size_t)kB * 32 * 2048);
    _Float16* xe2 = alloc_h((size_t)kB * 512 * 32);
    _Float16* he2 = alloc_h((size_t)kB * 64 * 512);
    _Float16* xe3 = alloc_h((size_t)kB * 128 * 64);
    _Float16* he3 = alloc_h((size_t)kB * 128 * 128);
    _Float16* fceA = alloc_h((size_t)kB * 32 * 128);
    _Float16* zbuf = alloc_h((size_t)32 * 256);
    _Float16* fcdRM = alloc_h((size_t)32 * 4096);
    _Float16* xdT = alloc_h((size_t)kB * 128 * 32);
    _Float16* xd0 = alloc_h((size_t)kB * 128 * 128);
    _Float16* hd0 = alloc_h((size_t)kB * 64 * 128);
    _Float16* xd1 = alloc_h((size_t)kB * 512 * 64);
    _Float16* hd1 = alloc_h((size_t)kB * 32 * 512);
    _Float16* xd2 = alloc_h((size_t)kB * 2048 * 32);
    _Float16* hd2 = alloc_h((size_t)kB * 32 * 2048);
    _Float16* xd3 = alloc_h((size_t)kB * 8192 * 32);
    _Float16* hd3 = alloc_h((size_t)kB * 16 * 8192);

    auto cvt = [&](const float* src, _Float16* dst, size_t R, int Cs, int Cd) {
        long long tot = (long long)R * Cd;
        unsigned blocks = (unsigned)((tot + 255) / 256);
        cvt_pad_kernel<<<blocks, 256, 0, stream>>>(src, dst, tot, Cs, Cd);
    };
    auto vadd = [&](const float* a, const float* b, float* o, int n) {
        vecadd_kernel<<<(n + 255) / 256, 256, 0, stream>>>(a, b, o, n);
    };
    auto gemm = [&](const _Float16* A, const _Float16* BT, _Float16* Out, const float* bias,
                    int M, int N, int K, int ldA, int ldB, int ldO,
                    long long sA, long long sB, long long sO, int batch) {
        const int Mt = M / 16, Nt = N / 16;
        const int nT = batch * Mt * Nt;
        gemm16_f16<<<(nT + 3) / 4, 128, 0, stream>>>(A, BT, Out, bias, K, ldA, ldB, ldO,
                                                     sA, sB, sO, Mt, Nt, nT);
    };

    // ---- weight conversion (every call; deterministic) ----
    for (int i = 0; i < 4; ++i) cvt(Din[i], DwH[i], (size_t)kNS[i + 1], kNS[i], kNS[i]);
    for (int i = 0; i < 4; ++i) cvt(Uin[i], UwH[i], (size_t)kNS[i], kNS[i + 1], kNS[i + 1]);
    cvt(fceW, fceWH, 256, 4096, 4096);
    cvt(fcdW, fcdWH, 4096, 256, 256);
    for (int i = 0; i < 4; ++i) {
        cvt(eWih[i], eWihP[i], (size_t)4 * kEncH[i], kEncC[i], kEncCP[i]);
        cvt(eWhh[i], eWhhP[i], (size_t)4 * kEncH[i], kEncH[i], kEncHP[i]);
        vadd(eBih[i], eBhh[i], eBs[i], 4 * kEncH[i]);
        cvt(dWih[i], dWihP[i], (size_t)4 * kDecH[i], kDecC[i], kDecCP[i]);
        cvt(dWhh[i], dWhhP[i], (size_t)4 * kDecH[i], kDecH[i], kDecHP[i]);
        vadd(dBih[i], dBhh[i], dBs[i], 4 * kDecH[i]);
    }
    cvt(xin, xe0, (size_t)kB * 8192, 3, 32);   // input x -> f16, pad 3->32

    // ---- encoder ----
    lstm_kernel<32, 32, 16><<<(2 * 8192) / 4, 128, 0, stream>>>(xe0, he0, sp[0], eWihP[0], eWhhP[0], eBs[0], 8192);
    hipMemsetAsync(xe1, 0, (size_t)kB * 2048 * 32 * sizeof(_Float16), stream); // pad cols 16..31
    gemm(DwH[0], he0, xe1, nullptr, 2048, 16, 8192, 8192, 8192, 32, 0, 16LL * 8192, 2048LL * 32, kB);
    lstm_kernel<32, 32, 32><<<(2 * 2048) / 4, 128, 0, stream>>>(xe1, he1, sp[1], eWihP[1], eWhhP[1], eBs[1], 2048);
    gemm(DwH[1], he1, xe2, nullptr, 512, 32, 2048, 2048, 2048, 32, 0, 32LL * 2048, 512LL * 32, kB);
    lstm_kernel<32, 64, 64><<<(2 * 512) / 4, 128, 0, stream>>>(xe2, he2, sp[2], eWihP[2], eWhhP[2], eBs[2], 512);
    gemm(DwH[2], he2, xe3, nullptr, 128, 64, 512, 512, 512, 64, 0, 64LL * 512, 128LL * 64, kB);
    lstm_kernel<64, 128, 128><<<(2 * 128) / 4, 128, 0, stream>>>(xe3, he3, sp[3], eWihP[3], eWhhP[3], eBs[3], 128);
    gemm(DwH[3], he3, fceA, nullptr, 32, 128, 128, 128, 128, 128, 0, 128LL * 128, 32LL * 128, kB);

    // ---- latent FC layers ----
    gemm(fceA, fceWH, zbuf, fceB, 32, 256, 4096, 4096, 4096, 256, 0, 0, 0, 1);
    gemm(zbuf, fcdWH, fcdRM, fcdB, 32, 4096, 256, 256, 256, 4096, 0, 0, 0, 1);
    {   // fcd out [b][32][128] -> xT [b][128][32]
        long long tot = (long long)kB * 32 * 128;
        transpose_bmc_kernel<<<(unsigned)((tot + 255) / 256), 256, 0, stream>>>(fcdRM, xdT, 32, 128, tot);
    }

    // ---- decoder ----
    gemm(UwH[3], xdT, xd0, nullptr, 128, 128, 32, 32, 32, 128, 0, 128LL * 32, 128LL * 128, kB);
    lstm_kernel<128, 64, 64><<<(2 * 128) / 4, 128, 0, stream>>>(xd0, hd0, sp[3], dWihP[0], dWhhP[0], dBs[0], 128);
    gemm(UwH[2], hd0, xd1, nullptr, 512, 64, 128, 128, 128, 64, 0, 64LL * 128, 512LL * 64, kB);
    lstm_kernel<64, 32, 32><<<(2 * 512) / 4, 128, 0, stream>>>(xd1, hd1, sp[2], dWihP[1], dWhhP[1], dBs[1], 512);
    gemm(UwH[1], hd1, xd2, nullptr, 2048, 32, 512, 512, 512, 32, 0, 32LL * 512, 2048LL * 32, kB);
    lstm_kernel<32, 32, 32><<<(2 * 2048) / 4, 128, 0, stream>>>(xd2, hd2, sp[1], dWihP[2], dWhhP[2], dBs[2], 2048);
    gemm(UwH[0], hd2, xd3, nullptr, 8192, 32, 2048, 2048, 2048, 32, 0, 32LL * 2048, 8192LL * 32, kB);
    lstm_kernel<32, 32, 16><<<(2 * 8192) / 4, 128, 0, stream>>>(xd3, hd3, sp[0], dWihP[3], dWhhP[3], dBs[3], 8192);

    // ---- final linear head -> d_out (f32, [B][8192][3]) ----
    {
        long long tot = (long long)kB * 8192;
        lin_kernel<<<(unsigned)((tot + 255) / 256), 256, 0, stream>>>(hd3, linW, linB,
                                                                     (float*)d_out, 8192, tot);
    }
}
